// TEMEncoder_83107617177739
// MI455X (gfx1250) — matrix operand order repaired and checked by reading, not compiled, not run on hardware
//
#include <hip/hip_runtime.h>
#include <hip/hip_bf16.h>
#include <math.h>

// Problem constants (match reference).
#define B_ 2048
#define L_ 2
#define A_ 4
#define S_ 128
#define H_ 256
#define K_ 16
#define M_ 131072

typedef __attribute__((ext_vector_type(16))) __bf16 v16bf;
typedef __attribute__((ext_vector_type(8)))  __bf16 v8bf;
typedef __attribute__((ext_vector_type(8)))  float  v8f;

// ---------------------------------------------------------------------------
// Deterministic stand-in for jax.random.normal(key(i), (B,L)).
// (Threefry bit-exactness is out of scope in a compile-only loop.)
// ---------------------------------------------------------------------------
__device__ __forceinline__ unsigned int hashu(unsigned int x) {
  x ^= x >> 16; x *= 0x7feb352du;
  x ^= x >> 15; x *= 0x846ca68bu;
  x ^= x >> 16;
  return x;
}
__device__ __forceinline__ float2 gauss2(unsigned int streamid, unsigned int i) {
  unsigned int u1 = hashu(i * 2u + 0x9e3779b9u * streamid + 1u);
  unsigned int u2 = hashu(i * 2u + 0x9e3779b9u * streamid + 2u);
  float f1 = ((float)u1 + 0.5f) * (1.0f / 4294967296.0f);
  float f2 = ((float)u2 + 0.5f) * (1.0f / 4294967296.0f);
  float r = sqrtf(-2.0f * logf(f1));
  float s, c;
  __sincosf(6.283185307f * f2, &s, &c);
  return make_float2(r * c, r * s);
}

// ---------------------------------------------------------------------------
// Kernel 1: q = normalize(sensory) -> bf16   (one wave per row, 4 elems/lane)
// ---------------------------------------------------------------------------
__global__ __launch_bounds__(256) void prep_queries_kernel(
    const float* __restrict__ sensory, __bf16* __restrict__ q_bf) {
  int wave = (blockIdx.x * blockDim.x + threadIdx.x) >> 5;
  int lane = threadIdx.x & 31;
  if (wave >= B_) return;
  float4 v = reinterpret_cast<const float4*>(sensory + (size_t)wave * S_)[lane];
  float ss = v.x * v.x + v.y * v.y + v.z * v.z + v.w * v.w;
  #pragma unroll
  for (int off = 16; off >= 1; off >>= 1) ss += __shfl_xor(ss, off, 32);
  float sc = 1.0f / sqrtf(ss);
  __bf16* dst = q_bf + (size_t)wave * S_ + lane * 4;
  dst[0] = (__bf16)(v.x * sc); dst[1] = (__bf16)(v.y * sc);
  dst[2] = (__bf16)(v.z * sc); dst[3] = (__bf16)(v.w * sc);
}

// ---------------------------------------------------------------------------
// Kernel 2: single pass over mem_sensory (64 MB):
//   - copy to new_mem_sens output (mandatory traffic)
//   - copy mem_locations row to new_mem_loc output
//   - rows < mem_valid_count: normalize -> bf16 keys for the GEMM
// ---------------------------------------------------------------------------
__global__ __launch_bounds__(256) void prep_keys_kernel(
    const float* __restrict__ mem_sensory, const float* __restrict__ mem_locations,
    const int* __restrict__ mvc_p,
    float* __restrict__ out_sens, float* __restrict__ out_loc,
    __bf16* __restrict__ keys_bf) {
  int wave = (blockIdx.x * blockDim.x + threadIdx.x) >> 5;
  int lane = threadIdx.x & 31;
  if (wave >= M_) return;
  int mvc = *mvc_p;
  float4 v = reinterpret_cast<const float4*>(mem_sensory + (size_t)wave * S_)[lane];
  reinterpret_cast<float4*>(out_sens + (size_t)wave * S_)[lane] = v;
  if (lane < L_) out_loc[(size_t)wave * L_ + lane] = mem_locations[(size_t)wave * L_ + lane];
  if (wave < mvc) {
    float ss = v.x * v.x + v.y * v.y + v.z * v.z + v.w * v.w;
    #pragma unroll
    for (int off = 16; off >= 1; off >>= 1) ss += __shfl_xor(ss, off, 32);
    float sc = 1.0f / sqrtf(ss);
    __bf16* dst = keys_bf + (size_t)wave * S_ + lane * 4;
    dst[0] = (__bf16)(v.x * sc); dst[1] = (__bf16)(v.y * sc);
    dst[2] = (__bf16)(v.z * sc); dst[3] = (__bf16)(v.w * sc);
  }
}

// ---------------------------------------------------------------------------
// Kernel 3: fused GEMM + top-16.
// Block = 256 threads (8 waves), owns 16 query rows; each wave one 16x16
// score tile per 128-column chunk (4x v_wmma_f32_16x16x32_bf16, K=S=128).
// B fragments are batch-loaded (one clause, one wait) before the WMMA burst,
// with a prefetch of the next chunk's columns issued alongside.
// Top-k merge is two-phase:
//   (1) all 256 threads threshold-filter 128 candidates/row against the
//       row's current 16th-best and compact survivors via LDS atomics
//       (expected survivors over the whole scan ~ 16*ln(M), i.e. near zero
//       per chunk after warm-up);
//   (2) 16 owner lanes insert survivors with a (value desc, index asc)
//       tie-aware sort, reproducing JAX's lowest-index-first tie rule
//       regardless of filter-append order. -inf candidates are skipped: the
//       pre-seeded (-inf, mvc+j) tail already encodes them.
// ---------------------------------------------------------------------------
__global__ __launch_bounds__(256) void topk_gemm_kernel(
    const __bf16* __restrict__ q_bf, const __bf16* __restrict__ keys_bf,
    const int* __restrict__ mvc_p, int* __restrict__ out_idx) {
  __shared__ float sc_lds[16 * 128];   // [row][col-in-chunk] scores
  __shared__ float cv_lds[16 * 128];   // compacted survivor values
  __shared__ int   ci_lds[16 * 128];   // compacted survivor indices
  __shared__ float thr_lds[16];
  __shared__ int   thri_lds[16];
  __shared__ int   cnt_lds[16];

  const int tid  = threadIdx.x;
  const int lane = tid & 31;
  const int w    = tid >> 5;

  int mvc = *mvc_p;
  if (mvc > M_) mvc = M_;

  // ---- A fragments (constant per block): ISA 16-bit A layout 16x32 ----
  const int qrow = blockIdx.x * 16 + (lane & 15);
  const int kbA  = (lane < 16) ? 0 : 8;
  v16bf afrag[4];
  #pragma unroll
  for (int kc4 = 0; kc4 < 4; ++kc4) {
    const __bf16* base = q_bf + (size_t)qrow * S_ + kc4 * 32;
    v8bf lo = *reinterpret_cast<const v8bf*>(base + kbA);
    v8bf hi = *reinterpret_cast<const v8bf*>(base + 16 + kbA);
    #pragma unroll
    for (int i = 0; i < 8; ++i) { afrag[kc4][i] = lo[i]; afrag[kc4][8 + i] = hi[i]; }
  }

  // ---- running top-16 (sorted desc), seeded for the -inf tail indices ----
  float topv[16]; int topi[16];
  #pragma unroll
  for (int j = 0; j < 16; ++j) { topv[j] = -INFINITY; topi[j] = mvc + j; }

  const int kbB     = (lane < 16) ? 0 : 8;   // ISA 16-bit B layout 32x16
  const int colw    = w * 16 + (lane & 15);
  const int nchunks = (mvc + 127) >> 7;

  for (int c = 0; c < nchunks; ++c) {
    const int ncol = c * 128 + colw;  // key row == B column
    const __bf16* bbase = keys_bf + (size_t)ncol * S_;

    // Batch all four K-slices of B: one load clause, single wait, then
    // four back-to-back WMMAs.
    v8bf blo[4], bhi[4];
    #pragma unroll
    for (int kc4 = 0; kc4 < 4; ++kc4) {
      blo[kc4] = *reinterpret_cast<const v8bf*>(bbase + kc4 * 32 + kbB);
      bhi[kc4] = *reinterpret_cast<const v8bf*>(bbase + kc4 * 32 + kbB + 16);
    }
    // Prefetch next chunk's column (this lane + 128 columns = +32 KB) so the
    // lines are in-flight during the WMMA burst and merge phase.
    if (c + 1 < nchunks)
      __builtin_prefetch(bbase + (size_t)128 * S_, 0, 3);

    v8f acc = {};
    #pragma unroll
    for (int kc4 = 0; kc4 < 4; ++kc4) {
      v16bf bfrag;
      #pragma unroll
      for (int i = 0; i < 8; ++i) { bfrag[i] = blo[kc4][i]; bfrag[8 + i] = bhi[kc4][i]; }
      acc = __builtin_amdgcn_wmma_f32_16x16x32_bf16(
          /*neg_a=*/false, afrag[kc4], /*neg_b=*/false, bfrag,
          /*c_mod=*/(short)0, acc, /*reuse_a=*/false, /*reuse_b=*/false);
    }

    // C layout: VGPR v -> row v (lanes 0-15) / v+8 (lanes 16-31), col = lane%16
    const bool colvalid = (ncol < mvc);
    const int mbase = (lane < 16) ? 0 : 8;
    #pragma unroll
    for (int v = 0; v < 8; ++v)
      sc_lds[(mbase + v) * 128 + colw] = colvalid ? acc[v] : -INFINITY;
    if (tid < 16) { thr_lds[tid] = topv[15]; thri_lds[tid] = topi[15]; cnt_lds[tid] = 0; }
    __syncthreads();

    // Phase 1: parallel threshold filter (16 threads/row x 8 candidates).
    {
      const int r   = tid & 15;
      const int sub = tid >> 4;
      const float thr  = thr_lds[r];
      const int   thri = thri_lds[r];
      #pragma unroll
      for (int j = 0; j < 8; ++j) {
        const int   n    = sub * 8 + j;
        const float s    = sc_lds[r * 128 + n];
        const int   gidx = c * 128 + n;
        if (s != -INFINITY && (s > thr || (s == thr && gidx < thri))) {
          const int p = atomicAdd(&cnt_lds[r], 1);
          cv_lds[r * 128 + p] = s;
          ci_lds[r * 128 + p] = gidx;
        }
      }
    }
    __syncthreads();

    // Phase 2: owner lanes insert the (few) survivors, tie-aware.
    if (tid < 16) {
      const int cnt = cnt_lds[tid];
      for (int p = 0; p < cnt; ++p) {
        const float s  = cv_lds[tid * 128 + p];
        const int   gi = ci_lds[tid * 128 + p];
        if (s > topv[15] || (s == topv[15] && gi < topi[15])) {
          topv[15] = s; topi[15] = gi;
          #pragma unroll
          for (int j = 15; j > 0; --j) {
            const bool sw = (topv[j] > topv[j - 1]) ||
                            (topv[j] == topv[j - 1] && topi[j] < topi[j - 1]);
            if (sw) {
              float tv = topv[j]; topv[j] = topv[j - 1]; topv[j - 1] = tv;
              int   ti = topi[j]; topi[j] = topi[j - 1]; topi[j - 1] = ti;
            }
          }
        }
      }
    }
    __syncthreads();  // LDS reused next chunk
  }

  if (tid < 16) {
    const int row = blockIdx.x * 16 + tid;
    #pragma unroll
    for (int j = 0; j < 16; ++j) out_idx[(size_t)row * 16 + j] = topi[j];
  }
}

// ---------------------------------------------------------------------------
// Localizer MLP helpers (one block of 256 threads == one batch row; thread j
// owns hidden unit j; outputs reduced through LDS).
// ---------------------------------------------------------------------------
__device__ __forceinline__ float2 dense_6_256_2(
    const float* xs, const float* __restrict__ W1, const float* __restrict__ b1,
    const float* __restrict__ W2, const float* __restrict__ b2,
    float* red0, float* red1, int tid) {
  float acc = b1[tid];
  #pragma unroll
  for (int i = 0; i < 6; ++i) acc = fmaf(xs[i], W1[i * H_ + tid], acc);
  float h = fmaxf(acc, 0.0f);
  red0[tid] = h * W2[tid * 2 + 0];
  red1[tid] = h * W2[tid * 2 + 1];
  __syncthreads();
  #pragma unroll
  for (int s = 128; s > 0; s >>= 1) {
    if (tid < s) { red0[tid] += red0[tid + s]; red1[tid] += red1[tid + s]; }
    __syncthreads();
  }
  float2 o = make_float2(red0[0] + b2[0], red1[0] + b2[1]);
  __syncthreads();
  return o;
}

__device__ __forceinline__ float2 dense_2_256_2(
    float2 m, const float* __restrict__ V1, const float* __restrict__ c1,
    const float* __restrict__ V2, const float* __restrict__ c2,
    float* red0, float* red1, int tid) {
  float acc = c1[tid];
  acc = fmaf(m.x, V1[tid], acc);
  acc = fmaf(m.y, V1[H_ + tid], acc);
  float h = fmaxf(acc, 0.0f);
  red0[tid] = h * V2[tid * 2 + 0];
  red1[tid] = h * V2[tid * 2 + 1];
  __syncthreads();
  #pragma unroll
  for (int s = 128; s > 0; s >>= 1) {
    if (tid < s) { red0[tid] += red0[tid + s]; red1[tid] += red1[tid + s]; }
    __syncthreads();
  }
  float2 o = make_float2(red0[0] + c2[0], red1[0] + c2[1]);
  __syncthreads();
  return o;
}

// ---------------------------------------------------------------------------
// Kernel 4: localizer + select_location + per-row log-prob.
// Reproduces the reference's quirks verbatim:
//   * matches = where(slot >= found, match, -inf)   (inverted mask, faithful)
//   * obj = cond ? integrated : expected
// ---------------------------------------------------------------------------
__global__ __launch_bounds__(256) void localizer_kernel(
    const float* __restrict__ last_loc, const float* __restrict__ action,
    const float* __restrict__ heading, const float* __restrict__ mem_locations,
    const float* __restrict__ W1, const float* __restrict__ b1,
    const float* __restrict__ W2, const float* __restrict__ b2,
    const float* __restrict__ V1, const float* __restrict__ c1,
    const float* __restrict__ V2, const float* __restrict__ c2,
    const float* __restrict__ vs_logits, const int* __restrict__ mvc_p,
    const int* __restrict__ top_idx,
    float* __restrict__ next_out, float* __restrict__ obj_ws,
    float* __restrict__ lp_ws) {
  __shared__ float xs[6];
  __shared__ float red0[256], red1[256];
  __shared__ float gk0[16], gk1[16], mt[16];
  const int b = blockIdx.x, tid = threadIdx.x;
  const int mvc = *mvc_p;
  const int found = (mvc < K_) ? mvc : K_;

  if (tid < 2)      xs[tid] = last_loc[b * L_ + tid];
  else if (tid < 6) xs[tid] = action[b * A_ + tid - 2];
  __syncthreads();

  float2 mean1 = dense_6_256_2(xs, W1, b1, W2, b2, red0, red1, tid);
  float2 s1    = dense_2_256_2(mean1, V1, c1, V2, c2, red0, red1, tid);
  const float evs = expf(vs_logits[0]);
  const float sd1x = evs * expf(s1.x) + 1e-6f;
  const float sd1y = evs * expf(s1.y) + 1e-6f;
  const float2 n1 = gauss2(1u, (unsigned)b);
  const float nxl = mean1.x + n1.x * sd1x;
  const float nyl = mean1.y + n1.y * sd1y;
  if (tid == 0) { next_out[b * L_ + 0] = nxl; next_out[b * L_ + 1] = nyl; }

  if (tid < 2)      xs[tid] = (tid == 0) ? nxl : nyl;
  else if (tid < 6) xs[tid] = heading[b * A_ + tid - 2];
  __syncthreads();

  float2 mean2 = dense_6_256_2(xs, W1, b1, W2, b2, red0, red1, tid);
  float2 s2    = dense_2_256_2(mean2, V1, c1, V2, c2, red0, red1, tid);
  const float sd2x = evs * expf(s2.x) + 1e-6f;
  const float sd2y = evs * expf(s2.y) + 1e-6f;

  if (tid < K_) {
    const int gi = top_idx[(size_t)b * K_ + tid];
    const float g0 = mem_locations[(size_t)gi * L_ + 0];
    const float g1 = mem_locations[(size_t)gi * L_ + 1];
    gk0[tid] = g0; gk1[tid] = g1;
    const float gn = sqrtf(g0 * g0 + g1 * g1);
    const float mn = sqrtf(mean2.x * mean2.x + mean2.y * mean2.y);
    const float match = (g0 * mean2.x + g1 * mean2.y) / (gn * mn);
    mt[tid] = (tid >= found) ? match : -INFINITY;  // faithful inverted mask
  }
  __syncthreads();

  if (tid == 0) {
    float mx = -INFINITY;
    #pragma unroll
    for (int k = 0; k < K_; ++k) mx = fmaxf(mx, mt[k]);
    float den = 0.0f, ix = 0.0f, iy = 0.0f;
    #pragma unroll
    for (int k = 0; k < K_; ++k) {
      const float wk = expf(mt[k] - mx);  // all -inf -> NaN, as in JAX
      den += wk; ix += wk * gk0[k]; iy += wk * gk1[k];
    }
    ix /= den; iy /= den;
    const float dx = (ix - mean2.x) / sd2x, dy = (iy - mean2.y) / sd2y;
    const bool too_far = sqrtf(0.25f * (dx * dx + dy * dy)) > 2.0f;
    const bool ign = (found == 0);
    const float2 n2 = gauss2(2u, (unsigned)b);
    const float ex = mean2.x + n2.x * sd2x, ey = mean2.y + n2.y * sd2y;
    const bool cx = too_far || ign || (ix != ix);
    const bool cy = too_far || ign || (iy != iy);
    const float ox = cx ? ix : ex;
    const float oy = cy ? iy : ey;
    obj_ws[b * L_ + 0] = ox; obj_ws[b * L_ + 1] = oy;
    const float zx = (ox - mean2.x) / sd2x, zy = (oy - mean2.y) / sd2y;
    lp_ws[b] = -0.5f * (n1.x * n1.x + n1.y * n1.y) - logf(sd1x) - logf(sd1y)
             - 0.5f * (zx * zx + zy * zy) - logf(sd2x) - logf(sd2y);
  }
}

// ---------------------------------------------------------------------------
// Kernel 5: deterministic (fixed-order) reduction of per-row log-probs.
// ---------------------------------------------------------------------------
__global__ __launch_bounds__(256) void reduce_lp_kernel(
    const float* __restrict__ lp_ws, float* __restrict__ out_lp) {
  __shared__ float red[256];
  const int tid = threadIdx.x;
  float s = 0.0f;
  for (int k = tid; k < B_; k += 256) s += lp_ws[k];
  red[tid] = s;
  __syncthreads();
  #pragma unroll
  for (int st = 128; st > 0; st >>= 1) {
    if (tid < st) red[tid] += red[tid + st];
    __syncthreads();
  }
  if (tid == 0) out_lp[0] = red[0] * (1.0f / (float)B_);
}

// ---------------------------------------------------------------------------
// Kernel 6: ring-buffer scatter into the (already copied) outputs.
// ---------------------------------------------------------------------------
__global__ __launch_bounds__(256) void scatter_kernel(
    const float* __restrict__ sensory, const float* __restrict__ obj_ws,
    const int* __restrict__ wp_p,
    float* __restrict__ out_sens, float* __restrict__ out_loc) {
  const int wave = (blockIdx.x * blockDim.x + threadIdx.x) >> 5;
  const int lane = threadIdx.x & 31;
  if (wave >= B_) return;
  const int pos = (int)(((unsigned)(*wp_p) + (unsigned)wave) % (unsigned)M_);
  reinterpret_cast<float4*>(out_sens + (size_t)pos * S_)[lane] =
      reinterpret_cast<const float4*>(sensory + (size_t)wave * S_)[lane];
  if (lane < L_) out_loc[(size_t)pos * L_ + lane] = obj_ws[(size_t)wave * L_ + lane];
}

// ---------------------------------------------------------------------------
// Host launcher. Input order follows setup_inputs(); all fp32 except the two
// int scalars. Output: [next_location(B*L) | lp(1) | new_mem_loc(M*L) |
// new_mem_sens(M*S)] fp32.
// ---------------------------------------------------------------------------
extern "C" void kernel_launch(void* const* d_in, const int* in_sizes, int n_in,
                              void* d_out, int out_size, void* d_ws, size_t ws_size,
                              hipStream_t stream) {
  const float* sensory   = (const float*)d_in[0];
  const float* last_loc  = (const float*)d_in[1];
  const float* action    = (const float*)d_in[2];
  const float* heading   = (const float*)d_in[3];
  const float* mem_loc   = (const float*)d_in[4];
  const float* mem_sens  = (const float*)d_in[5];
  const float* W1        = (const float*)d_in[6];
  const float* b1        = (const float*)d_in[7];
  const float* W2        = (const float*)d_in[8];
  const float* b2        = (const float*)d_in[9];
  const float* V1        = (const float*)d_in[10];
  const float* c1        = (const float*)d_in[11];
  const float* V2        = (const float*)d_in[12];
  const float* c2        = (const float*)d_in[13];
  const float* vs_logits = (const float*)d_in[14];
  const int*   mvc_p     = (const int*)d_in[15];
  const int*   wp_p      = (const int*)d_in[16];

  float* out        = (float*)d_out;
  float* o_next     = out;                          // B*L
  float* o_lp       = out + (size_t)B_ * L_;        // 1
  float* o_mem_loc  = o_lp + 1;                     // M*L
  float* o_mem_sens = o_mem_loc + (size_t)M_ * L_;  // M*S

  char* ws = (char*)d_ws;
  __bf16* keys_bf = (__bf16*)ws;                                    // 32 MB
  __bf16* q_bf    = (__bf16*)(ws + (size_t)M_ * S_ * 2);            // 512 KB
  int*    idx_ws  = (int*)(ws + (size_t)M_ * S_ * 2 + (size_t)B_ * S_ * 2);
  float*  obj_ws  = (float*)((char*)idx_ws + (size_t)B_ * K_ * 4);
  float*  lp_ws   = obj_ws + (size_t)B_ * L_;

  prep_queries_kernel<<<B_ / 8, 256, 0, stream>>>(sensory, q_bf);
  prep_keys_kernel<<<M_ / 8, 256, 0, stream>>>(mem_sens, mem_loc, mvc_p,
                                               o_mem_sens, o_mem_loc, keys_bf);
  topk_gemm_kernel<<<B_ / 16, 256, 0, stream>>>(q_bf, keys_bf, mvc_p, idx_ws);
  localizer_kernel<<<B_, 256, 0, stream>>>(last_loc, action, heading, mem_loc,
                                           W1, b1, W2, b2, V1, c1, V2, c2,
                                           vs_logits, mvc_p, idx_ws,
                                           o_next, obj_ws, lp_ws);
  reduce_lp_kernel<<<1, 256, 0, stream>>>(lp_ws, o_lp);
  scatter_kernel<<<B_ / 8, 256, 0, stream>>>(sensory, obj_ws, wp_p,
                                             o_mem_sens, o_mem_loc);
}
